// DGGCN_60722247631313
// MI455X (gfx1250) — compile-verified, matrix operand order checked by reading, not run on hardware
//
#include <hip/hip_runtime.h>
#include <hip/hip_bf16.h>

// ---------------------------------------------------------------------------
// Dual-direction gated GCN (2 layers) for MI455X / gfx1250, wave32 + WMMA.
//  N=50000 nodes, D=128 features, E=800000 edges.
//  GEMMs use V_WMMA_F32_16X16X32_BF16 with 3-term hi/lo bf16 split (~f32 acc).
//  Edge aggregation uses float4 gathers + global_atomic_add_f32 scatter.
// ---------------------------------------------------------------------------

#define N_NODES 50000
#define DIM     128
#define N_EDGES 800000
#define ROW_TILES (N_NODES / 16)   // 3125 (exact)

typedef __bf16 bf16_t;
typedef __attribute__((ext_vector_type(16))) __bf16 v16bf;
typedef __attribute__((ext_vector_type(8)))  float  v8f;

// ---- bf16 helpers via bit manipulation (round-to-nearest-even) -------------
__device__ __forceinline__ bf16_t f2bf(float f) {
    unsigned u = __builtin_bit_cast(unsigned, f);
    unsigned r = u + 0x7FFFu + ((u >> 16) & 1u);
    unsigned short s = (unsigned short)(r >> 16);
    return __builtin_bit_cast(bf16_t, s);
}
__device__ __forceinline__ float bf2f(bf16_t h) {
    unsigned short s = __builtin_bit_cast(unsigned short, h);
    unsigned u = ((unsigned)s) << 16;
    return __builtin_bit_cast(float, u);
}

__device__ __forceinline__ v8f wmma_bf(v16bf a, v16bf b, v8f c) {
    return __builtin_amdgcn_wmma_f32_16x16x32_bf16(
        /*neg_a=*/false, a, /*neg_b=*/false, b,
        /*c_mod=*/(short)0, c, /*reuse_a=*/false, /*reuse_b=*/false);
}

// Convert 16 f32 (already in A-layout order) to hi/lo bf16 vectors.
__device__ __forceinline__ void cvt_hilo(const float* f, v16bf& hi, v16bf& lo) {
#pragma unroll
    for (int i = 0; i < 16; ++i) {
        bf16_t h = f2bf(f[i]);
        hi[i] = h;
        lo[i] = f2bf(f[i] - bf2f(h));
    }
}

// Load lane's A fragment (row-major f32): two contiguous 8-float runs.
// 16-bit A 16x32 layout: lane = (khalf<<4)|row16; VGPR0..3 = K[kh*8 .. kh*8+7],
// VGPR4..7 = K[16+kh*8 .. 16+kh*8+7]  (K pairs packed per VGPR).
__device__ __forceinline__ void load_a_f32(const float* __restrict__ base,
                                           int arow, int kb, int khalf,
                                           float* f) {
    const float* ap = base + (size_t)arow * DIM + kb * 32 + khalf * 8;
    float4 q0 = *(const float4*)(ap);
    float4 q1 = *(const float4*)(ap + 4);
    float4 q2 = *(const float4*)(ap + 16);
    float4 q3 = *(const float4*)(ap + 20);
    f[0]=q0.x; f[1]=q0.y; f[2]=q0.z; f[3]=q0.w;
    f[4]=q1.x; f[5]=q1.y; f[6]=q1.z; f[7]=q1.w;
    f[8]=q2.x; f[9]=q2.y; f[10]=q2.z; f[11]=q2.w;
    f[12]=q3.x; f[13]=q3.y; f[14]=q3.z; f[15]=q3.w;
}

// ---- degree / normalization kernels ----------------------------------------
__global__ void k_deg_init(float* c1, float* c2) {
    int i = blockIdx.x * blockDim.x + threadIdx.x;
    if (i < N_NODES) { c1[i] = 1.0f; c2[i] = 1.0f; }   // +1 self-loop
}

__global__ void k_deg_count(const int* __restrict__ ei, float* c1, float* c2) {
    int e = blockIdx.x * blockDim.x + threadIdx.x;
    if (e < N_EDGES) {
        int s = ei[e];             // src row
        int d = ei[N_EDGES + e];   // dst row
        atomicAdd(&c1[d], 1.0f);   // in-degree  (forward conv)
        atomicAdd(&c2[s], 1.0f);   // out-degree (reversed conv)
    }
}

__global__ void k_deg_rsqrt(float* c1, float* c2) {
    int i = blockIdx.x * blockDim.x + threadIdx.x;
    if (i < N_NODES) { c1[i] = rsqrtf(c1[i]); c2[i] = rsqrtf(c2[i]); }
}

// ---- weight packing into WMMA B layout (hi/lo bf16) ------------------------
// B is KxN; per-lane: col = lane&15, khalf = lane>>4, 16 K values per lane
// in the same run structure as A. Packed index: ((nt*4+kb)*32+lane)*16 + i.
__global__ void k_pack_w(const float* __restrict__ W, bf16_t* __restrict__ hi,
                         bf16_t* __restrict__ lo, int transpose) {
    int t = blockIdx.x * 256 + threadIdx.x;   // 16384 elements total
    int i    = t & 15;
    int lane = (t >> 4) & 31;
    int kb   = (t >> 9) & 3;
    int nt   = t >> 11;
    int n     = nt * 16 + (lane & 15);
    int khalf = lane >> 4;
    int k     = kb * 32 + khalf * 8 + ((i < 8) ? i : (i - 8) + 16);
    float v = transpose ? W[n * DIM + k] : W[k * DIM + n];
    bf16_t h = f2bf(v);
    hi[t] = h;
    lo[t] = f2bf(v - bf2f(h));
}

// ---- feature GEMM: H = X @ W; agg1/agg2 = H*dinv^2 + bias (self-loop) ------
__global__ void __launch_bounds__(256)
k_gemm_feat(const float* __restrict__ X,
            const bf16_t* __restrict__ Bhi, const bf16_t* __restrict__ Blo,
            const float* __restrict__ bc,
            const float* __restrict__ d1, const float* __restrict__ d2,
            float* __restrict__ H, float* __restrict__ agg1,
            float* __restrict__ agg2) {
    int wave = threadIdx.x >> 5;
    int lane = threadIdx.x & 31;
    int tile = blockIdx.x * 8 + wave;
    if (tile >= ROW_TILES) return;               // wave-uniform
    int row0  = tile * 16;
    int arow  = row0 + (lane & 15);
    int khalf = lane >> 4;

    v8f acc[8];
#pragma unroll
    for (int nt = 0; nt < 8; ++nt) acc[nt] = (v8f){0,0,0,0,0,0,0,0};

#pragma unroll
    for (int kb = 0; kb < 4; ++kb) {
        float f[16];
        load_a_f32(X, arow, kb, khalf, f);
        v16bf ahi, alo;
        cvt_hilo(f, ahi, alo);
#pragma unroll
        for (int nt = 0; nt < 8; ++nt) {
            size_t boff = ((size_t)((nt * 4 + kb) * 32 + lane)) * 16;
            v16bf bhi = *(const v16bf*)(Bhi + boff);
            v16bf blo = *(const v16bf*)(Blo + boff);
            acc[nt] = wmma_bf(ahi, bhi, acc[nt]);
            acc[nt] = wmma_bf(ahi, blo, acc[nt]);
            acc[nt] = wmma_bf(alo, bhi, acc[nt]);
        }
    }

    int ncol = lane & 15;
#pragma unroll
    for (int nt = 0; nt < 8; ++nt) {
        int n = nt * 16 + ncol;
        float bcv = bc[n];
#pragma unroll
        for (int r = 0; r < 8; ++r) {
            int m = row0 + r + 8 * khalf;
            float v  = acc[nt][r];
            float e1 = d1[m], e2 = d2[m];
            size_t idx = (size_t)m * DIM + n;
            H[idx]    = v;
            agg1[idx] = v * e1 * e1 + bcv;
            agg2[idx] = v * e2 * e2 + bcv;
        }
    }
}

// ---- edge scatter: both directions in one pass (one wave per edge) ---------
__global__ void __launch_bounds__(256)
k_scatter(const int* __restrict__ ei, const float* __restrict__ H,
          const float* __restrict__ d1, const float* __restrict__ d2,
          float* __restrict__ agg1, float* __restrict__ agg2) {
    int wave = threadIdx.x >> 5;
    int lane = threadIdx.x & 31;
    int e = blockIdx.x * 8 + wave;
    if (e >= N_EDGES) return;
    int s = ei[e];
    int d = ei[N_EDGES + e];
    float n1 = d1[s] * d1[d];   // forward  norm
    float n2 = d2[s] * d2[d];   // reversed norm
    int c = lane * 4;
    float4 hs = *(const float4*)(H + (size_t)s * DIM + c);
    float4 hd = *(const float4*)(H + (size_t)d * DIM + c);
    float* p1 = agg1 + (size_t)d * DIM + c;
    float* p2 = agg2 + (size_t)s * DIM + c;
    atomicAdd(p1 + 0, hs.x * n1); atomicAdd(p1 + 1, hs.y * n1);
    atomicAdd(p1 + 2, hs.z * n1); atomicAdd(p1 + 3, hs.w * n1);
    atomicAdd(p2 + 0, hd.x * n2); atomicAdd(p2 + 1, hd.y * n2);
    atomicAdd(p2 + 2, hd.z * n2); atomicAdd(p2 + 3, hd.w * n2);
}

// ---- in-place ReLU on both aggregates --------------------------------------
__global__ void k_relu2(float* a, float* b) {
    size_t i = ((size_t)blockIdx.x * 256 + threadIdx.x) * 4;
    if (i < (size_t)N_NODES * DIM) {
        float4* pa = (float4*)(a + i);
        float4* pb = (float4*)(b + i);
        float4 va = *pa, vb = *pb;
        va.x = fmaxf(va.x, 0.f); va.y = fmaxf(va.y, 0.f);
        va.z = fmaxf(va.z, 0.f); va.w = fmaxf(va.w, 0.f);
        vb.x = fmaxf(vb.x, 0.f); vb.y = fmaxf(vb.y, 0.f);
        vb.z = fmaxf(vb.z, 0.f); vb.w = fmaxf(vb.w, 0.f);
        *pa = va; *pb = vb;
    }
}

// ---- fused gate: Z = o1@W1^T + o2@W2^T + b; out = sig(Z)*o1+(1-sig)*o2 -----
__global__ void __launch_bounds__(256)
k_gate(const float* __restrict__ o1, const float* __restrict__ o2,
       const bf16_t* __restrict__ B1hi, const bf16_t* __restrict__ B1lo,
       const bf16_t* __restrict__ B2hi, const bf16_t* __restrict__ B2lo,
       const float* __restrict__ bias, float* __restrict__ out) {
    int wave = threadIdx.x >> 5;
    int lane = threadIdx.x & 31;
    int tile = blockIdx.x * 8 + wave;
    if (tile >= ROW_TILES) return;               // wave-uniform
    int row0  = tile * 16;
    int arow  = row0 + (lane & 15);
    int khalf = lane >> 4;

    v8f acc[8];
#pragma unroll
    for (int nt = 0; nt < 8; ++nt) acc[nt] = (v8f){0,0,0,0,0,0,0,0};

#pragma unroll
    for (int kb = 0; kb < 4; ++kb) {
        float f1[16], f2[16];
        load_a_f32(o1, arow, kb, khalf, f1);
        load_a_f32(o2, arow, kb, khalf, f2);
        v16bf a1h, a1l, a2h, a2l;
        cvt_hilo(f1, a1h, a1l);
        cvt_hilo(f2, a2h, a2l);
#pragma unroll
        for (int nt = 0; nt < 8; ++nt) {
            size_t boff = ((size_t)((nt * 4 + kb) * 32 + lane)) * 16;
            v16bf b1h = *(const v16bf*)(B1hi + boff);
            v16bf b1l = *(const v16bf*)(B1lo + boff);
            v16bf b2h = *(const v16bf*)(B2hi + boff);
            v16bf b2l = *(const v16bf*)(B2lo + boff);
            acc[nt] = wmma_bf(a1h, b1h, acc[nt]);
            acc[nt] = wmma_bf(a1h, b1l, acc[nt]);
            acc[nt] = wmma_bf(a1l, b1h, acc[nt]);
            acc[nt] = wmma_bf(a2h, b2h, acc[nt]);
            acc[nt] = wmma_bf(a2h, b2l, acc[nt]);
            acc[nt] = wmma_bf(a2l, b2h, acc[nt]);
        }
    }

    int ncol = lane & 15;
#pragma unroll
    for (int nt = 0; nt < 8; ++nt) {
        int n = nt * 16 + ncol;
        float bv = bias[n];
#pragma unroll
        for (int r = 0; r < 8; ++r) {
            int m = row0 + r + 8 * khalf;
            size_t idx = (size_t)m * DIM + n;
            float z = acc[nt][r] + bv;
            float g = 1.0f / (1.0f + __expf(-z));
            out[idx] = g * o1[idx] + (1.0f - g) * o2[idx];
        }
    }
}

// ---------------------------------------------------------------------------
extern "C" void kernel_launch(void* const* d_in, const int* in_sizes, int n_in,
                              void* d_out, int out_size, void* d_ws,
                              size_t ws_size, hipStream_t stream) {
    const float* x   = (const float*)d_in[0];
    const int*   ei  = (const int*)d_in[1];
    const float* W1  = (const float*)d_in[2];
    const float* bc1 = (const float*)d_in[3];
    const float* W2  = (const float*)d_in[4];
    const float* bc2 = (const float*)d_in[5];
    const float* w11 = (const float*)d_in[6];
    const float* w12 = (const float*)d_in[7];
    const float* b1  = (const float*)d_in[8];
    const float* w21 = (const float*)d_in[9];
    const float* w22 = (const float*)d_in[10];
    const float* b2  = (const float*)d_in[11];
    float* out = (float*)d_out;

    // workspace layout (256B aligned except c2)
    char* ws = (char*)d_ws;
    const size_t feat = (size_t)N_NODES * DIM * sizeof(float);  // 25.6 MB
    float* H    = (float*)(ws + 0 * feat);
    float* Xn   = (float*)(ws + 1 * feat);
    float* agg1 = (float*)(ws + 2 * feat);
    float* agg2 = (float*)(ws + 3 * feat);
    const size_t wpk = (size_t)DIM * DIM;                       // elems / mat
    bf16_t* WP = (bf16_t*)(ws + 4 * feat);
    bf16_t* W1h = WP + 0 * wpk,  *W1l = WP + 1 * wpk;
    bf16_t* W2h = WP + 2 * wpk,  *W2l = WP + 3 * wpk;
    bf16_t* g11h = WP + 4 * wpk, *g11l = WP + 5 * wpk;
    bf16_t* g12h = WP + 6 * wpk, *g12l = WP + 7 * wpk;
    bf16_t* g21h = WP + 8 * wpk, *g21l = WP + 9 * wpk;
    bf16_t* g22h = WP + 10 * wpk, *g22l = WP + 11 * wpk;
    float* c1 = (float*)(ws + 4 * feat + 12 * wpk * sizeof(bf16_t));
    float* c2 = c1 + N_NODES;

    const int nodeBlocks = (N_NODES + 255) / 256;
    const int edgeBlocks = (N_EDGES + 255) / 256;
    const int gemmBlocks = (ROW_TILES + 7) / 8;    // 391
    const int scatBlocks = N_EDGES / 8;            // 100000 (exact)
    const int reluBlocks = (N_NODES * DIM) / (256 * 4);  // 6250 (exact)

    // normalization coefficients
    k_deg_init<<<nodeBlocks, 256, 0, stream>>>(c1, c2);
    k_deg_count<<<edgeBlocks, 256, 0, stream>>>(ei, c1, c2);
    k_deg_rsqrt<<<nodeBlocks, 256, 0, stream>>>(c1, c2);

    // pack weights (gate weights are used transposed: out = o @ w.T)
    k_pack_w<<<64, 256, 0, stream>>>(W1,  W1h,  W1l,  0);
    k_pack_w<<<64, 256, 0, stream>>>(W2,  W2h,  W2l,  0);
    k_pack_w<<<64, 256, 0, stream>>>(w11, g11h, g11l, 1);
    k_pack_w<<<64, 256, 0, stream>>>(w12, g12h, g12l, 1);
    k_pack_w<<<64, 256, 0, stream>>>(w21, g21h, g21l, 1);
    k_pack_w<<<64, 256, 0, stream>>>(w22, g22h, g22l, 1);

    // ---- layer 1 ----
    k_gemm_feat<<<gemmBlocks, 256, 0, stream>>>(x, W1h, W1l, bc1, c1, c2,
                                                H, agg1, agg2);
    k_scatter<<<scatBlocks, 256, 0, stream>>>(ei, H, c1, c2, agg1, agg2);
    k_relu2<<<reluBlocks, 256, 0, stream>>>(agg1, agg2);
    k_gate<<<gemmBlocks, 256, 0, stream>>>(agg1, agg2, g11h, g11l, g12h, g12l,
                                           b1, Xn);

    // ---- layer 2 ----
    k_gemm_feat<<<gemmBlocks, 256, 0, stream>>>(Xn, W2h, W2l, bc2, c1, c2,
                                                H, agg1, agg2);
    k_scatter<<<scatBlocks, 256, 0, stream>>>(ei, H, c1, c2, agg1, agg2);
    k_relu2<<<reluBlocks, 256, 0, stream>>>(agg1, agg2);
    k_gate<<<gemmBlocks, 256, 0, stream>>>(agg1, agg2, g21h, g21l, g22h, g22l,
                                           b2, out);
}
